// AdaptiveConv_45062796870258
// MI455X (gfx1250) — matrix operand order repaired and checked by reading, not compiled, not run on hardware
//
#include <hip/hip_runtime.h>
#include <stdint.h>

typedef __attribute__((ext_vector_type(16))) __bf16 v16bf;
typedef __attribute__((ext_vector_type(8)))  float  v8f;
typedef __attribute__((ext_vector_type(4)))  unsigned int v4u;
typedef __attribute__((ext_vector_type(8)))  int v8i;
typedef __attribute__((ext_vector_type(4)))  int v4i;

#define BB   8
#define TT   4
#define CIN  256
#define COUT 256
#define HH   64
#define WW   64
#define PW   66                 // padded width/height (1-px zero halo)
#define KTOT (CIN * 9)          // 2304
#define NCI  32                 // ci per stage
#define KC   (NCI * 9)          // 288 K per stage
#define NSTAGE (CIN / NCI)      // 8
#define NTILE  32               // pixels per block (N)
#define BSTRIDE 296             // shorts; rows 16B-aligned, bank-conflict-free
#define RAW_CI  102             // shorts per ci in raw patch (3*34)

__device__ __forceinline__ unsigned short f32_to_bf16(float f) {
  unsigned int u = __builtin_bit_cast(unsigned int, f);
  u += 0x7fffu + ((u >> 16) & 1u);
  return (unsigned short)(u >> 16);
}
__device__ __forceinline__ float bf16_to_f32(unsigned short h) {
  unsigned int u = ((unsigned int)h) << 16;
  return __builtin_bit_cast(float, u);
}

// ---- prep 1: split weights fp32 -> bf16 hi/lo planes (K-contiguous rows) ----
__global__ void wsplit_kernel(const float* __restrict__ w,
                              unsigned short* __restrict__ hi,
                              unsigned short* __restrict__ lo, int n) {
  int i = blockIdx.x * blockDim.x + threadIdx.x;
  if (i >= n) return;
  float x = w[i];
  unsigned short h = f32_to_bf16(x);
  hi[i] = h;
  lo[i] = f32_to_bf16(x - bf16_to_f32(h));
}

// ---- prep 2: split input fp32 -> zero-padded bf16 hi/lo planes [bt][ci][66][66] ----
__global__ void xsplit_kernel(const float* __restrict__ x,
                              unsigned short* __restrict__ hi,
                              unsigned short* __restrict__ lo) {
  size_t i = (size_t)blockIdx.x * 256 + threadIdx.x;
  const size_t total = (size_t)BB * TT * CIN * PW * PW;
  if (i >= total) return;
  int px = (int)(i % PW);
  size_t r = i / PW;
  int py = (int)(r % PW);
  size_t c = r / PW;                       // bt*CIN + ci
  float v = 0.0f;
  if (px >= 1 && px <= WW && py >= 1 && py <= HH)
    v = x[c * (HH * WW) + (size_t)(py - 1) * WW + (px - 1)];
  unsigned short h = f32_to_bf16(v);
  hi[i] = h;
  lo[i] = f32_to_bf16(v - bf16_to_f32(h));
}

// ---- main: implicit GEMM, TDM-staged raw patch, bf16x3 WMMA ----
__global__ __launch_bounds__(256)
void adaconv_kernel(const unsigned short* __restrict__ whi,
                    const unsigned short* __restrict__ wlo,
                    const unsigned short* __restrict__ xhi,
                    const unsigned short* __restrict__ xlo,
                    float* __restrict__ out) {
  __shared__ unsigned short rawHi[NCI * RAW_CI];   // TDM dest: [ci][3][34] bf16
  __shared__ unsigned short rawLo[NCI * RAW_CI];
  __shared__ unsigned short Bhi[NTILE * BSTRIDE];  // im2col [n][k] bf16
  __shared__ unsigned short Blo[NTILE * BSTRIDE];

  const int tid  = threadIdx.x;
  const int lane = tid & 31;
  const int wave = tid >> 5;
  const int half = lane >> 4;
  const int l16  = lane & 15;

  const int nt = blockIdx.x;          // 0..511 : (t, h, w0)
  const int cb = blockIdx.y;          // 0..1   : Cout/128
  const int b  = blockIdx.z;          // 0..7

  const int t  = nt >> 7;
  const int h  = (nt >> 1) & 63;
  const int w0 = (nt & 1) << 5;

  const int co = cb * 128 + wave * 16 + l16;       // A row (m = l16)
  const size_t wrow = (size_t)(b * COUT + co) * KTOT;

  // padded-plane element offset of tile start: row h (== gh=h-1 padded), col w0 (== gw=w0-1)
  const size_t xelem = ((size_t)(b * TT + t) * CIN) * (PW * PW) + (size_t)h * PW + w0;

  const unsigned ldsRawHi = (unsigned)(uintptr_t)(&rawHi[0]);
  const unsigned ldsRawLo = (unsigned)(uintptr_t)(&rawLo[0]);

  v8f acc0 = {};
  v8f acc1 = {};

  for (int kc = 0; kc < NSTAGE; ++kc) {
    // ---------- TDM: async 3D tile copy [32 ci][3 rows][34 cols] bf16, hi+lo ----------
    if (tid == 0) {
      const int ci0 = kc * NCI;
      unsigned long long gaHi =
          (unsigned long long)(uintptr_t)xhi + (xelem + (size_t)ci0 * (PW * PW)) * 2ull;
      unsigned long long gaLo =
          (unsigned long long)(uintptr_t)xlo + (xelem + (size_t)ci0 * (PW * PW)) * 2ull;

      v8i g1; v4i g2; v4i g3; v8i g4;
      g1[0] = 0x00010000;                 // data_size = 1 (2 bytes)
      g1[1] = 0;                          // tensor_dim0[15:0] = 0 (dim0 = 1<<20)
      g1[2] = 0x10;                       // tensor_dim0[31:16]=0x10; tensor_dim1[15:0]=0
      g1[3] = 0x10 | (34 << 16);          // tensor_dim1[31:16]=0x10; tile_dim0 = 34
      g1[4] = 3 | (NCI << 16);            // tile_dim1 = 3; tile_dim2 = 32
      g1[5] = PW;                         // tensor_dim0_stride = 66
      g1[6] = (PW * PW) << 16;            // tensor_dim1_stride[15:0] = 4356
      g1[7] = 0;                          // tensor_dim1_stride[47:16] = 0
      g2[0] = 1 << 20;                    // tensor_dim2 (large; tile always in-bounds)
      g2[1] = 1;                          // tensor_dim3
      g2[2] = 0; g2[3] = 0;               // dim2 stride / tile_dim3 unused
      g3[0] = 0; g3[1] = 0; g3[2] = 0; g3[3] = 0;
      g4[0] = 0; g4[1] = 0; g4[2] = 0; g4[3] = 0;
      g4[4] = 0; g4[5] = 0; g4[6] = 0; g4[7] = 0;

      v4u g0;
      g0[0] = 1u;                                               // count=1
      g0[1] = ldsRawHi;                                         // lds_addr
      g0[2] = (unsigned)(gaHi & 0xffffffffull);                 // global_addr lo
      g0[3] = (unsigned)((gaHi >> 32) & 0x01ffffffull) | (2u << 30);  // hi + type=2
      __builtin_amdgcn_tensor_load_to_lds(g0, g1, g2, g3, g4, 0);

      g0[1] = ldsRawLo;
      g0[2] = (unsigned)(gaLo & 0xffffffffull);
      g0[3] = (unsigned)((gaLo >> 32) & 0x01ffffffull) | (2u << 30);
      __builtin_amdgcn_tensor_load_to_lds(g0, g1, g2, g3, g4, 0);

      __builtin_amdgcn_s_wait_tensorcnt(0);
    }
    __syncthreads();

    // ---------- im2col expand raw patch -> B planes (pure bf16 moves) ----------
    #pragma unroll 4
    for (int i = 0; i < (NTILE * KC) / 256; ++i) {   // 36 per thread
      int e  = tid + i * 256;
      int n  = e / KC;
      int kl = e - n * KC;
      int ci  = kl / 9;
      int tap = kl - ci * 9;
      int kh  = tap / 3;
      int kw  = tap - kh * 3;
      int ra = ci * RAW_CI + kh * 34 + n + kw;       // raw col = n + kw
      Bhi[n * BSTRIDE + kl] = rawHi[ra];
      Blo[n * BSTRIDE + kl] = rawLo[ra];
    }
    __syncthreads();

    // ---------- 9 chunks of K=32; A reused over 2 N-subtiles; 6 WMMAs/chunk ----------
    #pragma unroll 3
    for (int c = 0; c < KC / 32; ++c) {
      const int k0 = kc * KC + c * 32;
      union { uint4 q[2]; v16bf v; } ah, al, bh0, bl0, bh1, bl1;

      ah.q[0] = *reinterpret_cast<const uint4*>(whi + wrow + k0 + half * 16);
      ah.q[1] = *reinterpret_cast<const uint4*>(whi + wrow + k0 + half * 16 + 8);
      al.q[0] = *reinterpret_cast<const uint4*>(wlo + wrow + k0 + half * 16);
      al.q[1] = *reinterpret_cast<const uint4*>(wlo + wrow + k0 + half * 16 + 8);

      const int boff = c * 32 + half * 16;
      const uint4* p0h = reinterpret_cast<const uint4*>(&Bhi[l16 * BSTRIDE + boff]);
      const uint4* p0l = reinterpret_cast<const uint4*>(&Blo[l16 * BSTRIDE + boff]);
      const uint4* p1h = reinterpret_cast<const uint4*>(&Bhi[(16 + l16) * BSTRIDE + boff]);
      const uint4* p1l = reinterpret_cast<const uint4*>(&Blo[(16 + l16) * BSTRIDE + boff]);
      bh0.q[0] = p0h[0]; bh0.q[1] = p0h[1];
      bl0.q[0] = p0l[0]; bl0.q[1] = p0l[1];
      bh1.q[0] = p1h[0]; bh1.q[1] = p1h[1];
      bl1.q[0] = p1l[0]; bl1.q[1] = p1l[1];

      acc0 = __builtin_amdgcn_wmma_f32_16x16x32_bf16(false, ah.v, false, bh0.v, (short)0, acc0, false, false);
      acc0 = __builtin_amdgcn_wmma_f32_16x16x32_bf16(false, al.v, false, bh0.v, (short)0, acc0, false, false);
      acc0 = __builtin_amdgcn_wmma_f32_16x16x32_bf16(false, ah.v, false, bl0.v, (short)0, acc0, false, false);
      acc1 = __builtin_amdgcn_wmma_f32_16x16x32_bf16(false, ah.v, false, bh1.v, (short)0, acc1, false, false);
      acc1 = __builtin_amdgcn_wmma_f32_16x16x32_bf16(false, al.v, false, bh1.v, (short)0, acc1, false, false);
      acc1 = __builtin_amdgcn_wmma_f32_16x16x32_bf16(false, ah.v, false, bl1.v, (short)0, acc1, false, false);
    }
    __syncthreads();
  }

  // ---- store C/D: VGPR r -> (M = r + half*8, N = l16 [+16]) ----
  float* op = out + (size_t)(b * TT + t) * COUT * (HH * WW) + (size_t)h * WW + w0;
  const int mbase = cb * 128 + wave * 16 + half * 8;
  #pragma unroll
  for (int r = 0; r < 8; ++r) {
    op[(size_t)(mbase + r) * (HH * WW) + l16]      = acc0[r];
    op[(size_t)(mbase + r) * (HH * WW) + 16 + l16] = acc1[r];
  }
}

extern "C" void kernel_launch(void* const* d_in, const int* in_sizes, int n_in,
                              void* d_out, int out_size, void* d_ws, size_t ws_size,
                              hipStream_t stream) {
  const float* x  = (const float*)d_in[0];   // [B*T, Cin, H, W] fp32
  const float* aw = (const float*)d_in[1];   // [B, Cout, Cin, 3, 3] fp32
  float* out = (float*)d_out;

  const int    NW = BB * COUT * KTOT;                       // 4,718,592
  const size_t NX = (size_t)BB * TT * CIN * PW * PW;        // 35,684,352

  unsigned short* whi = (unsigned short*)d_ws;
  unsigned short* wlo = whi + NW;
  unsigned short* xhi = wlo + NW;
  unsigned short* xlo = xhi + NX;                           // total ws ~162 MB

  wsplit_kernel<<<(NW + 255) / 256, 256, 0, stream>>>(aw, whi, wlo, NW);
  xsplit_kernel<<<(unsigned)((NX + 255) / 256), 256, 0, stream>>>(x, xhi, xlo);

  dim3 grid(512, 2, 8);                      // N tiles x Cout tiles x batch
  adaconv_kernel<<<grid, 256, 0, stream>>>(whi, wlo, xhi, xlo, out);
}
